// Gauss_re_481036337394
// MI455X (gfx1250) — compile-verified, matrix operand order checked by reading, not run on hardware
//
#include <hip/hip_runtime.h>
#include <hip/hip_bf16.h>
#include <math.h>

typedef _Float16 f16;
typedef __attribute__((ext_vector_type(16))) _Float16 v16h;
typedef __attribute__((ext_vector_type(8)))  _Float16 v8h;
typedef __attribute__((ext_vector_type(8)))  float    v8f;

#define BATCH 64
#define NPTS  512
#define HD    128
#define NINP  64
#define DY    8
#define AUGW  520   // 512 + 8 RHS columns
#define ALPHA 1e-5f

// Fragment layout for v_wmma_f32_16x16x32_f16 (CDNA5 ISA 7.12.2):
// lane L (h = L>>4) element i holds K = (i&7) + 8h + 16*(i>>3).
// => v16h = two contiguous 8-half (16-byte) runs at +8h and +16+8h.
__device__ __forceinline__ v16h load_frag(const f16* base, int hh) {
  union { v16h v; v8h h[2]; } u;
  u.h[0] = *(const v8h*)(base + 8 * hh);        // K = 0..7   (+8h)
  u.h[1] = *(const v8h*)(base + 16 + 8 * hh);   // K = 16..23 (+8h)
  return u.v;
}

// ---------------------------------------------------------------------------
// Weight pre-conversion: f32 -> f16 once per launch.
__global__ void cvt_weights(const float* __restrict__ W1,
                            const float* __restrict__ W2,
                            const float* __restrict__ W3,
                            f16* __restrict__ W1h, f16* __restrict__ W2h,
                            f16* __restrict__ W3h) {
  int t = blockIdx.x * blockDim.x + threadIdx.x;
  const int n1 = HD * NINP;        // 8192
  const int n2 = HD * HD;          // 16384
  if (t < n1) W1h[t] = (f16)W1[t];
  if (t < n2) { W2h[t] = (f16)W2[t]; W3h[t] = (f16)W3[t]; }
}

// ---------------------------------------------------------------------------
// One Linear layer on a 16-row tile: out[16][128] = in[16][Kdim] @ W^T + b.
__device__ __forceinline__ void mlp_layer(const f16* sIn, int ldin,
                                          const f16* __restrict__ W,
                                          const float* __restrict__ bias,
                                          int Kdim, bool relu,
                                          f16* sOut, int ldout,
                                          float* fOut, int ldf, int lane) {
  const int m  = lane & 15;
  const int hh = lane >> 4;
  for (int nt = 0; nt < 8; ++nt) {
    v8f acc = {};
    for (int k0 = 0; k0 < Kdim; k0 += 32) {
      v16h a   = load_frag(sIn + m * ldin + k0, hh);            // A row m
      v16h bfr = load_frag(W + (nt * 16 + m) * Kdim + k0, hh);  // B col n = W row n
      acc = __builtin_amdgcn_wmma_f32_16x16x32_f16(
          false, a, false, bfr, (short)0, acc, false, false);
    }
    float bv = bias[nt * 16 + m];
#pragma unroll
    for (int r = 0; r < 8; ++r) {
      float v = acc[r] + bv;
      if (relu) v = fmaxf(v, 0.0f);
      int row = r + (hh << 3);
      int col = nt * 16 + m;
      if (sOut) sOut[row * ldout + col] = (f16)v;
      if (fOut) fOut[row * ldf + col] = v;
    }
  }
}

// grid = 2 * 64 * 32 one-wave blocks; each embeds 16 rows of x or x2.
__global__ __launch_bounds__(32) void embed_kernel(
    const float* __restrict__ x, const float* __restrict__ x2,
    const f16* __restrict__ W1h, const float* __restrict__ b1,
    const f16* __restrict__ W2h, const float* __restrict__ b2,
    const f16* __restrict__ W3h, const float* __restrict__ b3,
    f16* __restrict__ e, f16* __restrict__ e2) {
  __shared__ __align__(16) f16   sA[16 * 136];   // 136 halves = 272 B = 17*16 B
  __shared__ __align__(16) f16   sB[16 * 136];
  __shared__ float sH[16 * 132];
  __shared__ float sInv[16];

  int gb    = blockIdx.x;
  int which = gb >> 11;          // 0: x -> e, 1: x2 -> e2
  int rem   = gb & 2047;
  int b     = rem >> 5;
  int rt    = rem & 31;
  int lane  = threadIdx.x;
  int rowbase = rt * 16;

  const float* X = which ? x2 : x;
  for (int idx = lane; idx < 16 * NINP; idx += 32) {
    int r = idx >> 6, c = idx & 63;
    sA[r * 136 + c] = (f16)X[((size_t)(b * NPTS + rowbase + r)) * NINP + c];
  }
  __syncthreads();
  mlp_layer(sA, 136, W1h, b1, NINP, false, sB, 136, nullptr, 0, lane);
  __syncthreads();
  mlp_layer(sB, 136, W2h, b2, HD, true, sA, 136, nullptr, 0, lane);  // ReLU after 2nd linear
  __syncthreads();
  mlp_layer(sA, 136, W3h, b3, HD, false, nullptr, 0, sH, 132, lane);
  __syncthreads();

  if (lane < 16) {
    float ss = 0.0f;
    for (int c = 0; c < HD; ++c) { float v = sH[lane * 132 + c]; ss += v * v; }
    sInv[lane] = 1.0f / fmaxf(sqrtf(ss), 1e-12f);
  }
  __syncthreads();

  f16* E = which ? e2 : e;
  for (int idx = lane; idx < 16 * HD; idx += 32) {
    int r = idx >> 7, c = idx & 127;
    E[((size_t)(b * NPTS + rowbase + r)) * HD + c] = (f16)(sH[r * 132 + c] * sInv[r]);
  }
}

// ---------------------------------------------------------------------------
// Gram kernel: 32x32 output tile per wave (2x2 WMMA register tile).
// mode 0: K = |e e^T| + alpha*I  -> augmented system (cols 0..511)
// mode 1: K_ = |e2 e^T|          -> scratch
__global__ __launch_bounds__(32) void gram_kernel(
    const f16* __restrict__ e, const f16* __restrict__ e2,
    float* __restrict__ Aug, float* __restrict__ Kq) {
  int gb   = blockIdx.x;
  int mode = gb >> 14;           // 64*16*16 = 16384 tiles per mode
  int rem  = gb & 16383;
  int b    = rem >> 8;
  int qt   = (rem >> 4) & 15;
  int nt   = rem & 15;
  int lane = threadIdx.x;
  int m = lane & 15, hh = lane >> 4;

  const f16* Arows = (mode ? e2 : e) + (size_t)b * NPTS * HD;
  const f16* Brows = e + (size_t)b * NPTS * HD;

  v8f acc00 = {}, acc01 = {}, acc10 = {}, acc11 = {};
  for (int k0 = 0; k0 < HD; k0 += 32) {
    v16h a0 = load_frag(Arows + (qt * 32 + m) * HD + k0, hh);
    v16h a1 = load_frag(Arows + (qt * 32 + 16 + m) * HD + k0, hh);
    v16h b0 = load_frag(Brows + (nt * 32 + m) * HD + k0, hh);       // e^T cols
    v16h b1 = load_frag(Brows + (nt * 32 + 16 + m) * HD + k0, hh);
    acc00 = __builtin_amdgcn_wmma_f32_16x16x32_f16(false, a0, false, b0, (short)0, acc00, false, false);
    acc01 = __builtin_amdgcn_wmma_f32_16x16x32_f16(false, a0, false, b1, (short)0, acc01, false, false);
    acc10 = __builtin_amdgcn_wmma_f32_16x16x32_f16(false, a1, false, b0, (short)0, acc10, false, false);
    acc11 = __builtin_amdgcn_wmma_f32_16x16x32_f16(false, a1, false, b1, (short)0, acc11, false, false);
  }

#pragma unroll
  for (int ti = 0; ti < 2; ++ti) {
#pragma unroll
    for (int tj = 0; tj < 2; ++tj) {
      const v8f& acc = ti ? (tj ? acc11 : acc10) : (tj ? acc01 : acc00);
      int col = nt * 32 + tj * 16 + m;
#pragma unroll
      for (int r = 0; r < 8; ++r) {
        int row = qt * 32 + ti * 16 + r + (hh << 3);
        float v = fabsf(acc[r]);                    // sqrt(K^2) == |K|
        if (mode == 0) {
          if (row == col) v += ALPHA;
          Aug[((size_t)b * NPTS + row) * AUGW + col] = v;
        } else {
          Kq[((size_t)b * NPTS + row) * NPTS + col] = v;
        }
      }
    }
  }
}

__global__ void fill_y(const float* __restrict__ y, float* __restrict__ Aug) {
  int t = blockIdx.x * blockDim.x + threadIdx.x;
  if (t >= BATCH * NPTS * DY) return;
  int d = t & 7;
  int i = (t >> 3) & (NPTS - 1);
  int b = t >> 12;
  Aug[((size_t)b * NPTS + i) * AUGW + NPTS + d] = y[t];
}

// ---------------------------------------------------------------------------
// Gauss-Jordan on [K+aI | y] per batch; one 1024-thread block (32 waves) per
// batch. Pivot row staged in LDS; next pivot row prefetched during the
// elimination phase (global_prefetch_b8).
__global__ __launch_bounds__(1024) void gj_solve(float* __restrict__ Aug) {
  float* A = Aug + (size_t)blockIdx.x * NPTS * AUGW;
  __shared__ float prow[AUGW];
  __shared__ float pinv_s;
  int tid  = threadIdx.x;
  int wave = tid >> 5;

  for (int k = 0; k < NPTS; ++k) {
    if (tid == 0) pinv_s = 1.0f / A[(size_t)k * AUGW + k];
    __syncthreads();
    float pinv = pinv_s;
    for (int j = k + tid; j < AUGW; j += 1024) {
      float v = A[(size_t)k * AUGW + j] * pinv;
      A[(size_t)k * AUGW + j] = v;
      prow[j] = v;
    }
    __syncthreads();
    // Prefetch next pivot row while eliminating (33 cachelines).
    if (k + 1 < NPTS && tid < 33)
      __builtin_prefetch(&A[(size_t)(k + 1) * AUGW + k + tid * 16], 0, 1);
    for (int i = wave; i < NPTS; i += 32) {
      if (i == k) continue;
      float f = A[(size_t)i * AUGW + k];
      for (int j = k + 1 + (tid & 31); j < AUGW; j += 32)
        A[(size_t)i * AUGW + j] = fmaf(-f, prow[j], A[(size_t)i * AUGW + j]);
    }
    __syncthreads();
  }
}

// out[b][q][d] = sum_n K_[b][q][n] * Z[b][n][d]; Z lives in Aug cols 512..519.
__global__ __launch_bounds__(256) void final_kernel(
    const float* __restrict__ Kq, const float* __restrict__ Aug,
    float* __restrict__ out) {
  int b  = blockIdx.x >> 4;
  int qc = blockIdx.x & 15;
  int tid = threadIdx.x;
  int q = qc * 32 + (tid >> 3);
  int d = tid & 7;
  const float* Krow = Kq + ((size_t)b * NPTS + q) * NPTS;
  const float* Zcol = Aug + (size_t)b * NPTS * AUGW + NPTS + d;
  float acc = 0.0f;
  for (int n = 0; n < NPTS; ++n)
    acc = fmaf(Krow[n], Zcol[(size_t)n * AUGW], acc);
  out[((size_t)b * NPTS + q) * DY + d] = acc;
}

extern "C" void kernel_launch(void* const* d_in, const int* in_sizes, int n_in,
                              void* d_out, int out_size, void* d_ws, size_t ws_size,
                              hipStream_t stream) {
  const float* x  = (const float*)d_in[0];
  const float* x2 = (const float*)d_in[1];
  const float* y  = (const float*)d_in[2];
  const float* W1 = (const float*)d_in[3];
  const float* b1 = (const float*)d_in[4];
  const float* W2 = (const float*)d_in[5];
  const float* b2 = (const float*)d_in[6];
  const float* W3 = (const float*)d_in[7];
  const float* b3 = (const float*)d_in[8];
  float* out = (float*)d_out;

  // Workspace layout (bytes):
  //   e   : f16 [64][512][128]   =  8 MiB @ 0
  //   e2  : f16 [64][512][128]   =  8 MiB @ 8 MiB
  //   K_  : f32 [64][512][512]   = 64 MiB @ 16 MiB
  //   Aug : f32 [64][512][520]   = 65 MiB @ 80 MiB
  //   W1h/W2h/W3h : f16          = 80 KiB @ 145 MiB
  char* ws = (char*)d_ws;
  f16*   e   = (f16*)(ws);
  f16*   e2  = (f16*)(ws + (size_t)8 * 1024 * 1024);
  float* Kq  = (float*)(ws + (size_t)16 * 1024 * 1024);
  float* Aug = (float*)(ws + (size_t)80 * 1024 * 1024);
  f16*   W1h = (f16*)(ws + (size_t)145 * 1024 * 1024);
  f16*   W2h = W1h + HD * NINP;
  f16*   W3h = W2h + HD * HD;

  cvt_weights<<<(HD * HD + 255) / 256, 256, 0, stream>>>(W1, W2, W3, W1h, W2h, W3h);
  embed_kernel<<<2 * BATCH * (NPTS / 16), 32, 0, stream>>>(
      x, x2, W1h, b1, W2h, b2, W3h, b3, e, e2);
  gram_kernel<<<2 * BATCH * (NPTS / 32) * (NPTS / 32), 32, 0, stream>>>(
      e, e2, Aug, Kq);
  fill_y<<<(BATCH * NPTS * DY + 255) / 256, 256, 0, stream>>>(y, Aug);
  gj_solve<<<BATCH, 1024, 0, stream>>>(Aug);
  final_kernel<<<BATCH * 16, 256, 0, stream>>>(Kq, Aug, out);
}
